// Mamba_40415642255481
// MI455X (gfx1250) — matrix operand order repaired and checked by reading, not compiled, not run on hardware
//
#include <hip/hip_runtime.h>
#include <hip/hip_bf16.h>
#include <math.h>

#define D_MODEL 1024
#define D_INNER 2048
#define D_STATE 16
#define D_CONV 4
#define DT_RANK 64
#define BATCH 4
#define SEQLEN 2048
#define BL (BATCH * SEQLEN) /* 8192 token rows */

typedef __bf16 bf16;
typedef __attribute__((ext_vector_type(16))) __bf16 v16bf;
typedef __attribute__((ext_vector_type(8)))  __bf16 v8bf;
typedef __attribute__((ext_vector_type(8)))  float  v8f;

// ---------------------------------------------------------------------------
// WMMA fragment loaders (layouts per CDNA5 ISA §7.12.2, wave32)
// A 16x32 bf16: lane l<16 -> row l, elems 0..7 = K k0..k0+7, elems 8..15 = K k0+16..k0+23
//               lane l+16 -> row l, K halves shifted by +8
template <int LDA>
__device__ __forceinline__ v16bf load_a_frag(const bf16* __restrict__ A,
                                             int row0, int k0, int lane) {
    const int hi = lane >> 4;
    const bf16* p = A + (size_t)(row0 + (lane & 15)) * LDA + k0 + hi * 8;
    union { v16bf v; v8bf h[2]; } u;
    u.h[0] = *(const v8bf*)(p);        // K = k0 + hi*8 .. +7
    u.h[1] = *(const v8bf*)(p + 16);   // K = k0 + 16 + hi*8 .. +7
    return u.v;
}

// B 32x16 bf16 (column n = weight row n, K contiguous):
// lanes 0..15 -> col n, elems 0..7 = K k0..k0+7, elems 8..15 = K k0+16..k0+23 ; lanes 16..31 -> col n-16, K halves shifted by +8
template <int LDW>
__device__ __forceinline__ v16bf load_b_frag(const bf16* __restrict__ W,
                                             int col0, int k0, int lane) {
    const int hi = lane >> 4;
    const bf16* p = W + (size_t)(col0 + (lane & 15)) * LDW + k0 + hi * 8;
    union { v16bf v; v8bf h[2]; } u;
    u.h[0] = *(const v8bf*)(p);        // K = k0 + hi*8 .. +7
    u.h[1] = *(const v8bf*)(p + 16);   // K = k0 + 16 + hi*8 .. +7
    return u.v;
}

__device__ __forceinline__ float softplusf(float x) {
    return (x > 20.f) ? x : log1pf(__expf(x));
}

// ---------------------------------------------------------------------------
// bf16 WMMA GEMM: C[M x N] = A[M x K] * W[N x K]^T (f32 accumulate)
// Block tile 128x128 (8 waves, wave tile 32x64), double-buffered k-pipeline.
// Compile-time N,K -> immediate-offset addressing. M%128==0, N%128==0, K%64==0.
template <int N, int K, bool SOFTPLUS>
__global__ __launch_bounds__(256)
void gemm_bf16_wmma(const bf16* __restrict__ A, const bf16* __restrict__ W,
                    float* __restrict__ C, const float* __restrict__ bias) {
    const int lane = threadIdx.x & 31;
    const int wave = threadIdx.x >> 5;
    const int row0 = blockIdx.y * 128 + (wave >> 1) * 32; // 4 waves along M
    const int col0 = blockIdx.x * 128 + (wave & 1) * 64;  // 2 waves along N

    const v8f zero8 = {0.f, 0.f, 0.f, 0.f, 0.f, 0.f, 0.f, 0.f};
    v8f acc[2][4];
#pragma unroll
    for (int i = 0; i < 2; ++i)
#pragma unroll
        for (int j = 0; j < 4; ++j) acc[i][j] = zero8;

    v16bf a0[2], b0[4], a1[2], b1[4];

    auto load_set = [&](v16bf* a, v16bf* b, int k0) {
        a[0] = load_a_frag<K>(A, row0, k0, lane);
        a[1] = load_a_frag<K>(A, row0 + 16, k0, lane);
#pragma unroll
        for (int j = 0; j < 4; ++j) b[j] = load_b_frag<K>(W, col0 + 16 * j, k0, lane);
    };
    auto compute = [&](const v16bf* a, const v16bf* b) {
#pragma unroll
        for (int i = 0; i < 2; ++i)
#pragma unroll
            for (int j = 0; j < 4; ++j)
                acc[i][j] = __builtin_amdgcn_wmma_f32_16x16x32_bf16(
                    false, a[i], false, b[j], (short)0, acc[i][j], false, false);
    };

    // software pipeline, unrolled by 64 in K (all K here are multiples of 64)
    load_set(a0, b0, 0);
    int k0 = 0;
    for (; k0 < K - 64; k0 += 64) {
        load_set(a1, b1, k0 + 32);
        compute(a0, b0);
        load_set(a0, b0, k0 + 64);
        compute(a1, b1);
    }
    load_set(a1, b1, k0 + 32);
    compute(a0, b0);
    compute(a1, b1);

    // C/D layout: elem v -> row = row0 + 8*(lane>=16) + v (+16 per i), col = col0 + (lane&15) + 16*j
    const int r_base = row0 + ((lane >> 4) << 3);
    const int c_base = col0 + (lane & 15);
    float* p = C + (size_t)r_base * N + c_base;
#pragma unroll
    for (int i = 0; i < 2; ++i)
#pragma unroll
        for (int j = 0; j < 4; ++j) {
            const float bv = SOFTPLUS ? bias[c_base + 16 * j] : 0.f;
#pragma unroll
            for (int v = 0; v < 8; ++v) {
                float val = acc[i][j][v];
                if (SOFTPLUS) val = softplusf(val + bv);
                p[(size_t)(16 * i + v) * N + 16 * j] = val; // immediate offsets
            }
        }
}

// ---------------------------------------------------------------------------
// f32 -> bf16 convert with optional zero tail-padding (for x_proj_w 96->128 rows)
__global__ void cvt_f32_bf16(const float* __restrict__ src, bf16* __restrict__ dst,
                             size_t n_src, size_t n_dst) {
    size_t i = (size_t)blockIdx.x * blockDim.x + threadIdx.x;
    if (i < n_dst) dst[i] = (i < n_src) ? (bf16)src[i] : (bf16)0.f;
}

// Causal depthwise conv (width 4) + SiLU; reads x half of xz (ld = 2*D_INNER)
__global__ void conv_silu_kernel(const float* __restrict__ xz,
                                 const float* __restrict__ conv_w,
                                 const float* __restrict__ conv_b,
                                 float* __restrict__ xconv_f32,
                                 bf16* __restrict__ xconv_bf16) {
    size_t idx = (size_t)blockIdx.x * blockDim.x + threadIdx.x;
    if (idx >= (size_t)BL * D_INNER) return;
    const int d = (int)(idx % D_INNER);
    const size_t bl = idx / D_INNER;
    const int l = (int)(bl % SEQLEN);
    const size_t brow = bl - l; // b*SEQLEN
    float s = conv_b[d];
#pragma unroll
    for (int w = 0; w < D_CONV; ++w) {
        const int ls = l - (D_CONV - 1) + w;
        if (ls >= 0) s += xz[(brow + ls) * (2 * D_INNER) + d] * conv_w[d * D_CONV + w];
    }
    const float v = s / (1.f + __expf(-s)); // silu
    xconv_f32[idx] = v;
    xconv_bf16[idx] = (bf16)v;
}

// Extract dt_in (first 64 of padded 128-wide x_dbl) -> bf16 for dt_proj GEMM
__global__ void extract_dt_kernel(const float* __restrict__ xdbl, bf16* __restrict__ dt) {
    size_t i = (size_t)blockIdx.x * blockDim.x + threadIdx.x;
    if (i >= (size_t)BL * DT_RANK) return;
    const size_t r = i >> 6;
    const int c = (int)(i & 63);
    dt[i] = (bf16)xdbl[r * 128 + c];
}

// ---------------------------------------------------------------------------
// Selective scan: one lane per (b, d, n) -> 131072 lanes. Block = 1 batch x 16 d
// channels x 16 states = 256 threads. B/C/delta/u chunks staged in LDS; y reduced
// over n via shfl_xor; gating + D-skip + bf16 convert fused in epilogue.
__global__ __launch_bounds__(256)
void scan_kernel(const float* __restrict__ delta, // BL x D_INNER
                 const float* __restrict__ xdbl,  // BL x 128 (B @ +64, C @ +80)
                 const float* __restrict__ u,     // BL x D_INNER  (x_conv f32)
                 const float* __restrict__ xz,    // BL x 4096    (z @ +D_INNER)
                 const float* __restrict__ A_log, // D_INNER x 16
                 const float* __restrict__ Dp,    // D_INNER
                 bf16* __restrict__ yout) {       // BL x D_INNER
    const int b = blockIdx.y;
    const int d0 = blockIdx.x * 16;
    const int tid = threadIdx.x;
    const int dl = tid >> 4;   // local channel 0..15
    const int n = tid & 15;    // state index
    const int d = d0 + dl;
    const float a = -__expf(A_log[(size_t)d * D_STATE + n]);
    const float Dd = Dp[d];
    const size_t base = (size_t)b * SEQLEN;

    __shared__ float sB[64][16], sC[64][16], sDelta[64][16], sU[64][16];

    float h = 0.f;
    for (int l0 = 0; l0 < SEQLEN; l0 += 64) {
        __syncthreads();
#pragma unroll
        for (int i = 0; i < 4; ++i) {
            const int e = tid + i * 256;
            const int t = e >> 4, j = e & 15;
            const size_t row = base + l0 + t;
            sB[t][j] = xdbl[row * 128 + 64 + j];
            sC[t][j] = xdbl[row * 128 + 80 + j];
            sDelta[t][j] = delta[row * D_INNER + d0 + j];
            sU[t][j] = u[row * D_INNER + d0 + j];
        }
        __syncthreads();
        for (int t = 0; t < 64; ++t) {
            const float dlt = sDelta[t][dl];
            const float uu = sU[t][dl];
            h = h * __expf(dlt * a) + (dlt * uu) * sB[t][n];
            float pv = h * sC[t][n];
            pv += __shfl_xor(pv, 1);
            pv += __shfl_xor(pv, 2);
            pv += __shfl_xor(pv, 4);
            pv += __shfl_xor(pv, 8);
            if (n == 0) {
                const size_t row = base + l0 + t;
                const float zv = xz[row * (2 * D_INNER) + D_INNER + d];
                const float g = zv / (1.f + __expf(-zv)); // silu(z)
                yout[row * D_INNER + d] = (bf16)((pv + uu * Dd) * g);
            }
        }
    }
}

// ---------------------------------------------------------------------------
extern "C" void kernel_launch(void* const* d_in, const int* in_sizes, int n_in,
                              void* d_out, int out_size, void* d_ws, size_t ws_size,
                              hipStream_t stream) {
    (void)in_sizes; (void)n_in; (void)out_size; (void)ws_size;
    const float* hs     = (const float*)d_in[0]; // 4x2048x1024
    const float* w_in   = (const float*)d_in[1]; // 4096x1024
    const float* conv_w = (const float*)d_in[2]; // 2048x4
    const float* conv_b = (const float*)d_in[3]; // 2048
    const float* w_xp   = (const float*)d_in[4]; // 96x2048
    const float* w_dt   = (const float*)d_in[5]; // 2048x64
    const float* dt_b   = (const float*)d_in[6]; // 2048
    const float* A_log  = (const float*)d_in[7]; // 2048x16
    const float* Dp     = (const float*)d_in[8]; // 2048
    const float* w_out  = (const float*)d_in[9]; // 1024x2048
    float* out = (float*)d_out;                  // 4x2048x1024

    char* ws = (char*)d_ws;
    size_t off = 0;
    auto alloc = [&](size_t bytes) {
        char* p = ws + off;
        off += (bytes + 255) & ~(size_t)255;
        return p;
    };
    bf16* hs_bf     = (bf16*)alloc((size_t)BL * D_MODEL * 2);       // 16.8 MB
    bf16* w_in_bf   = (bf16*)alloc((size_t)4096 * 1024 * 2);        //  8.4 MB
    bf16* w_xp_bf   = (bf16*)alloc((size_t)128 * 2048 * 2);         //  0.5 MB (padded 96->128)
    bf16* w_dt_bf   = (bf16*)alloc((size_t)2048 * 64 * 2);          //  0.3 MB
    bf16* w_out_bf  = (bf16*)alloc((size_t)1024 * 2048 * 2);        //  4.2 MB
    float* xz       = (float*)alloc((size_t)BL * 4096 * 4);         // 134 MB
    float* xconv_f  = (float*)alloc((size_t)BL * D_INNER * 4);      //  67 MB
    bf16* xconv_bf  = (bf16*)alloc((size_t)BL * D_INNER * 2);       //  34 MB
    float* xdbl     = (float*)alloc((size_t)BL * 128 * 4);          //   4 MB (padded 96->128)
    bf16* dt_bf     = (bf16*)alloc((size_t)BL * DT_RANK * 2);       //   1 MB
    float* delta    = (float*)alloc((size_t)BL * D_INNER * 4);      //  67 MB
    bf16* y_bf      = (bf16*)alloc((size_t)BL * D_INNER * 2);       //  34 MB

    auto blocks = [](size_t n) { return (unsigned)((n + 255) / 256); };

    // 1) precision conversions
    cvt_f32_bf16<<<blocks((size_t)BL * D_MODEL), 256, 0, stream>>>(hs, hs_bf, (size_t)BL * D_MODEL, (size_t)BL * D_MODEL);
    cvt_f32_bf16<<<blocks(4096ull * 1024), 256, 0, stream>>>(w_in, w_in_bf, 4096ull * 1024, 4096ull * 1024);
    cvt_f32_bf16<<<blocks(128ull * 2048), 256, 0, stream>>>(w_xp, w_xp_bf, 96ull * 2048, 128ull * 2048);
    cvt_f32_bf16<<<blocks(2048ull * 64), 256, 0, stream>>>(w_dt, w_dt_bf, 2048ull * 64, 2048ull * 64);
    cvt_f32_bf16<<<blocks(1024ull * 2048), 256, 0, stream>>>(w_out, w_out_bf, 1024ull * 2048, 1024ull * 2048);

    // 2) in_proj: xz = hs @ in_proj_w^T   (8192 x 4096, K=1024)
    gemm_bf16_wmma<4096, 1024, false><<<dim3(4096 / 128, BL / 128), 256, 0, stream>>>(
        hs_bf, w_in_bf, xz, nullptr);

    // 3) causal conv + silu
    conv_silu_kernel<<<blocks((size_t)BL * D_INNER), 256, 0, stream>>>(
        xz, conv_w, conv_b, xconv_f, xconv_bf);

    // 4) x_proj: x_dbl = x_conv @ x_proj_w^T  (8192 x 128pad, K=2048)
    gemm_bf16_wmma<128, 2048, false><<<dim3(1, BL / 128), 256, 0, stream>>>(
        xconv_bf, w_xp_bf, xdbl, nullptr);

    // 5) dt_in -> bf16
    extract_dt_kernel<<<blocks((size_t)BL * DT_RANK), 256, 0, stream>>>(xdbl, dt_bf);

    // 6) dt_proj + bias + softplus: delta (8192 x 2048, K=64)
    gemm_bf16_wmma<2048, 64, true><<<dim3(2048 / 128, BL / 128), 256, 0, stream>>>(
        dt_bf, w_dt_bf, delta, dt_b);

    // 7) selective scan + D-skip + silu(z) gating -> y_bf16
    scan_kernel<<<dim3(D_INNER / 16, BATCH), 256, 0, stream>>>(
        delta, xdbl, xconv_f, xz, A_log, Dp, y_bf);

    // 8) out_proj: out = y @ out_proj_w^T  (8192 x 1024, K=2048)
    gemm_bf16_wmma<1024, 2048, false><<<dim3(1024 / 128, BL / 128), 256, 0, stream>>>(
        y_bf, w_out_bf, out, nullptr);
}
